// DeepSpeedStyleMoE_44925357916271
// MI455X (gfx1250) — matrix operand order repaired and checked by reading, not compiled, not run on hardware
//
#include <hip/hip_runtime.h>
#include <hip/hip_bf16.h>
#include <hip/hip_fp16.h>

#define NUM_EXPERTS 64
#define D_MODEL     1024
#define N_TOKENS    4096
#define CAPACITY    512
#define KSTEPS      (D_MODEL / 32)

typedef _Float16 v16h __attribute__((ext_vector_type(16)));
typedef _Float16 v8h  __attribute__((ext_vector_type(8)));
typedef float    v8f  __attribute__((ext_vector_type(8)));
typedef float    v4f  __attribute__((ext_vector_type(4)));
typedef unsigned int u32x4 __attribute__((ext_vector_type(4)));
typedef int      i32x4 __attribute__((ext_vector_type(4)));
typedef int      i32x8 __attribute__((ext_vector_type(8)));

union HFrag { v16h v; v8h h[2]; };

#if defined(__has_builtin)
#if __has_builtin(__builtin_amdgcn_tensor_load_to_lds) && __has_builtin(__builtin_amdgcn_s_wait_tensorcnt)
#define MOE_USE_TDM 1
#endif
#endif

// ---------------------------------------------------------------- zero init
__global__ __launch_bounds__(256) void moe_zero(float* __restrict__ out,
                                                int* __restrict__ cnt) {
  const size_t i = (size_t)blockIdx.x * 256 + threadIdx.x;
  out[i] = 0.0f;
  if (i < NUM_EXPERTS) cnt[i] = 0;
}

// ---------------------------------------------------------------- gating + dispatch
__device__ __forceinline__ bool pref(float v1, int i1, float v2, int i2) {
  return (v1 > v2) || (v1 == v2 && i1 < i2);
}

__global__ __launch_bounds__(128) void moe_gate_dispatch(const float* __restrict__ x,
                                                         const float* __restrict__ gw,
                                                         _Float16* __restrict__ buf,
                                                         int* __restrict__ cnt,
                                                         int* __restrict__ rtok,
                                                         float* __restrict__ rw) {
  const int t    = blockIdx.x * 4 + (threadIdx.x >> 5);   // one token per wave
  const int lane = threadIdx.x & 31;
  const float* xr  = x  + (size_t)t * D_MODEL;
  const float* g0p = gw + (size_t)lane * D_MODEL;
  const float* g1p = gw + (size_t)(lane + 32) * D_MODEL;
  float a0 = 0.f, a1 = 0.f;                               // 2 experts per lane
  for (int d = 0; d < D_MODEL; d += 4) {
    v4f xv = *(const v4f*)(xr + d);
    v4f w0 = *(const v4f*)(g0p + d);
    v4f w1 = *(const v4f*)(g1p + d);
#pragma unroll
    for (int i = 0; i < 4; ++i) { a0 = fmaf(xv[i], w0[i], a0); a1 = fmaf(xv[i], w1[i], a1); }
  }
  // local sorted pair, then butterfly top-2 merge across the wave
  float b1, b2; int i1, i2;
  if (pref(a0, lane, a1, lane + 32)) { b1 = a0; i1 = lane;      b2 = a1; i2 = lane + 32; }
  else                               { b1 = a1; i1 = lane + 32; b2 = a0; i2 = lane; }
  for (int off = 16; off; off >>= 1) {
    float c1 = __shfl_xor(b1, off), c2 = __shfl_xor(b2, off);
    int   j1 = __shfl_xor(i1, off), j2 = __shfl_xor(i2, off);
    float n1, n2; int m1, m2;
    if (pref(b1, i1, c1, j1)) {
      n1 = b1; m1 = i1;
      if (pref(b2, i2, c1, j1)) { n2 = b2; m2 = i2; } else { n2 = c1; m2 = j1; }
    } else {
      n1 = c1; m1 = j1;
      if (pref(c2, j2, b1, i1)) { n2 = c2; m2 = j2; } else { n2 = b1; m2 = i1; }
    }
    b1 = n1; i1 = m1; b2 = n2; i2 = m2;
  }
  // softmax over the two selected logits (b1 >= b2, stable)
  const float eneg = __expf(b2 - b1);
  const float w1 = 1.0f / (1.0f + eneg);
  const float w2 = 1.0f - w1;
  int s0 = 0, s1 = 0;
  if (lane == 0) {
    s0 = atomicAdd(&cnt[i1], 1);
    s1 = atomicAdd(&cnt[i2], 1);
  }
  s0 = __shfl(s0, 0); s1 = __shfl(s1, 0);
  if (s0 < CAPACITY) {
    _Float16* dst = buf + ((size_t)i1 * CAPACITY + s0) * D_MODEL;
    for (int d = lane; d < D_MODEL; d += 32) dst[d] = (_Float16)xr[d];
    if (lane == 0) { rtok[i1 * CAPACITY + s0] = t; rw[i1 * CAPACITY + s0] = w1; }
  }
  if (s1 < CAPACITY) {
    _Float16* dst = buf + ((size_t)i2 * CAPACITY + s1) * D_MODEL;
    for (int d = lane; d < D_MODEL; d += 32) dst[d] = (_Float16)xr[d];
    if (lane == 0) { rtok[i2 * CAPACITY + s1] = t; rw[i2 * CAPACITY + s1] = w2; }
  }
}

#if defined(MOE_USE_TDM)
// Issue a TDM load of a 128-row x 32-col fp32 tile (row stride D_MODEL) into LDS,
// padding 4 dwords after every 32 dwords -> 144B LDS row stride.
__device__ __forceinline__ void tdm_load_tile(const float* gptr, unsigned ldsOff) {
  const unsigned long long ga = (unsigned long long)(const void*)gptr;
  u32x4 g0;
  g0[0] = 1u;                                                // count=1, user desc
  g0[1] = ldsOff;                                            // lds_addr (bytes)
  g0[2] = (unsigned)ga;                                      // global_addr[31:0]
  g0[3] = ((unsigned)(ga >> 32) & 0x01FFFFFFu) | (2u << 30); // addr[56:32] | type=2
  i32x8 g1;
  g1[0] = (2 << 16) | (1 << 20) | (4 << 22) | (3 << 25);     // 4B elems, pad_en, 32dw interval, 4dw pad
  g1[1] = (int)(((unsigned)D_MODEL & 0xFFFFu) << 16);        // tensor_dim0 lo16
  g1[2] = (int)(((unsigned)D_MODEL >> 16) | (((unsigned)D_MODEL & 0xFFFFu) << 16));
  g1[3] = (int)(((unsigned)D_MODEL >> 16) | (32u << 16));    // dim1 hi | tile_dim0=32
  g1[4] = 128;                                               // tile_dim1=128, tile_dim2=0
  g1[5] = D_MODEL;                                           // tensor_dim0_stride lo32
  g1[6] = 0;
  g1[7] = 0;
  i32x4 g2 = {}; i32x4 g3 = {};
  i32x8 g4 = {};
  __builtin_amdgcn_tensor_load_to_lds(g0, g1, g2, g3, g4, 0);
}
#endif

// ---------------------------------------------------------------- expert GEMM + fused combine
// Block = 8 waves; computes a [<=256 rows] x [128 features] slab per pass.
// Double-buffered TDM stages fp32 W tiles into LDS; one cooperative pass converts
// to an f16 tile (80B row stride, conflict-free b128 reads) feeding the WMMAs.
__global__ __launch_bounds__(256) void moe_expert_gemm(const _Float16* __restrict__ buf,
                                                       const float* __restrict__ ew,
                                                       const int* __restrict__ cnt,
                                                       const int* __restrict__ rtok,
                                                       const float* __restrict__ rw,
                                                       float* __restrict__ out) {
  __shared__ float    Bt32[2][128 * 36];  // 2 x 18432 B fp32 staging (TDM dest)
  __shared__ _Float16 Bt16[128 * 40];     // 10240 B f16 tile (row stride 40 halves)
  const int e     = blockIdx.x >> 3;
  const int fbase = (blockIdx.x & 7) * 128;
  int cntE = cnt[e]; if (cntE > CAPACITY) cntE = CAPACITY;
  if (cntE == 0) return;
  const _Float16* bufE = buf + (size_t)e * CAPACITY * D_MODEL;
  const float*    We   = ew  + (size_t)e * D_MODEL * D_MODEL;
  const int wid  = threadIdx.x >> 5;
  const int lane = threadIdx.x & 31;
  const int lh   = lane >> 4;             // half-wave (K-split)
  const int ll   = lane & 15;
  // convert-pass coordinates: thread -> (row, 16-col half)
  const int cvr = threadIdx.x >> 1;
  const int cvc = (threadIdx.x & 1) << 4;

  for (int p = 0; p < CAPACITY; p += 256) {
    if (p >= cntE) break;                 // block-uniform
    const int mrow0 = p + wid * 16;
    const int mrow1 = p + (wid + 8) * 16;
    // wave-uniform activity flags, forced into SGPRs -> scalar branches.
    // Note act1 implies act0 (mrow1 > mrow0).
    const bool act0 = __builtin_amdgcn_readfirstlane(mrow0 < cntE ? 1 : 0) != 0;
    const bool act1 = __builtin_amdgcn_readfirstlane(mrow1 < cntE ? 1 : 0) != 0;
    v8f acc0[8], acc1[8];
#pragma unroll
    for (int i = 0; i < 8; ++i) { acc0[i] = v8f{}; acc1[i] = v8f{}; }

#if defined(MOE_USE_TDM)
    if (threadIdx.x < 32)
      tdm_load_tile(We + (size_t)fbase * D_MODEL,
                    (unsigned)(unsigned long long)(void*)&Bt32[0][0]);
#endif
    for (int ks = 0; ks < KSTEPS; ++ks) {
      const int k0 = ks * 32;
#if defined(MOE_USE_TDM)
      const int cb = ks & 1;
      if (threadIdx.x < 32) {
        __builtin_amdgcn_s_wait_tensorcnt(0);     // tile ks landed in Bt32[cb]
        if (ks + 1 < KSTEPS)                      // prefetch next tile, overlaps compute
          tdm_load_tile(We + (size_t)fbase * D_MODEL + (k0 + 32),
                        (unsigned)(unsigned long long)(void*)&Bt32[cb ^ 1][0]);
      }
      __syncthreads();                            // fp32 tile visible to all waves
      {                                           // cooperative fp32 -> f16 (once per block)
        const float* src = &Bt32[cb][cvr * 36 + cvc];
        v4f f0 = *(const v4f*)(src);
        v4f f1 = *(const v4f*)(src + 4);
        v4f f2 = *(const v4f*)(src + 8);
        v4f f3 = *(const v4f*)(src + 12);
        HFrag hv;
#pragma unroll
        for (int i = 0; i < 4; ++i) {
          hv.v[i]      = (_Float16)f0[i];
          hv.v[4 + i]  = (_Float16)f1[i];
          hv.v[8 + i]  = (_Float16)f2[i];
          hv.v[12 + i] = (_Float16)f3[i];
        }
        _Float16* dst = &Bt16[cvr * 40 + cvc];
        *(v8h*)dst = hv.h[0];
        *(v8h*)(dst + 8) = hv.h[1];
      }
      __syncthreads();                            // f16 tile ready
#else
      __syncthreads();
      for (int idx = threadIdx.x; idx < 128 * 32; idx += 256) {
        const int r = idx >> 5, c = idx & 31;
        Bt16[r * 40 + c] = (_Float16)We[(size_t)(fbase + r) * D_MODEL + k0 + c];
      }
      __syncthreads();
#endif

      // A fragments (16x32 f16; lanes<16 hold K 0-7/16-23, lanes>=16 hold K 8-15/24-31).
      // Activity test hoisted out of the N-tile loop: one scalar branch per K-step.
      if (act1) {                                 // both M tiles active
        HFrag a0, a1;
        const _Float16* ap0 = bufE + (size_t)(mrow0 + ll) * D_MODEL + k0 + (lh ? 8 : 0);
        a0.h[0] = *(const v8h*)ap0; a0.h[1] = *(const v8h*)(ap0 + 16);
        const _Float16* ap1 = bufE + (size_t)(mrow1 + ll) * D_MODEL + k0 + (lh ? 8 : 0);
        a1.h[0] = *(const v8h*)ap1; a1.h[1] = *(const v8h*)(ap1 + 16);
#pragma unroll
        for (int nt = 0; nt < 8; ++nt) {
          const _Float16* bp = &Bt16[(nt * 16 + ll) * 40 + (lh << 3)];
          HFrag bf;
          bf.h[0] = *(const v8h*)bp;
          bf.h[1] = *(const v8h*)(bp + 16);
          acc0[nt] = __builtin_amdgcn_wmma_f32_16x16x32_f16(false, a0.v, false, bf.v,
                                                            (short)0, acc0[nt], false, false);
          acc1[nt] = __builtin_amdgcn_wmma_f32_16x16x32_f16(false, a1.v, false, bf.v,
                                                            (short)0, acc1[nt], false, false);
        }
      } else if (act0) {                          // only first M tile active
        HFrag a0;
        const _Float16* ap0 = bufE + (size_t)(mrow0 + ll) * D_MODEL + k0 + (lh ? 8 : 0);
        a0.h[0] = *(const v8h*)ap0; a0.h[1] = *(const v8h*)(ap0 + 16);
#pragma unroll
        for (int nt = 0; nt < 8; ++nt) {
          const _Float16* bp = &Bt16[(nt * 16 + ll) * 40 + (lh << 3)];
          HFrag bf;
          bf.h[0] = *(const v8h*)bp;
          bf.h[1] = *(const v8h*)(bp + 16);
          acc0[nt] = __builtin_amdgcn_wmma_f32_16x16x32_f16(false, a0.v, false, bf.v,
                                                            (short)0, acc0[nt], false, false);
        }
      }
    }

    // Epilogue: fused combine. C layout: VGPR v, lanes 0-15 -> M=v, lanes 16-31 -> M=v+8.
#pragma unroll
    for (int half = 0; half < 2; ++half) {
      const int  mbase = half ? mrow1 : mrow0;
      const bool act   = half ? act1  : act0;
      if (!act) continue;
      v8f* acc = half ? acc1 : acc0;
#pragma unroll
      for (int v = 0; v < 8; ++v) {
        const int r = mbase + v + (lh << 3);
        if (r < cntE) {
          const int   tok = rtok[e * CAPACITY + r];
          const float w   = rw[e * CAPACITY + r];
          float* orow = out + (size_t)tok * D_MODEL + fbase + ll;
#pragma unroll
          for (int nt = 0; nt < 8; ++nt)
            unsafeAtomicAdd(orow + nt * 16, w * acc[nt][v]);
        }
      }
    }
  }
}

// ---------------------------------------------------------------- launch
extern "C" void kernel_launch(void* const* d_in, const int* in_sizes, int n_in,
                              void* d_out, int out_size, void* d_ws, size_t ws_size,
                              hipStream_t stream) {
  const float* x  = (const float*)d_in[0];   // [4096,1024]
  const float* gw = (const float*)d_in[1];   // [64,1024]
  const float* ew = (const float*)d_in[2];   // [64,1024,1024]
  float* out = (float*)d_out;                // [4096,1024]

  const size_t BUF_BYTES   = (size_t)NUM_EXPERTS * CAPACITY * D_MODEL * sizeof(_Float16); // 64 MB
  const size_t ROUTE_BYTES = (size_t)NUM_EXPERTS * CAPACITY * sizeof(int);                // 128 KB
  char* wsb = (char*)d_ws;
  _Float16* buf  = (_Float16*)wsb;
  int*      cnt  = (int*)  (wsb + BUF_BYTES);
  int*      rtok = (int*)  (wsb + BUF_BYTES + 4096);
  float*    rw   = (float*)(wsb + BUF_BYTES + 4096 + ROUTE_BYTES);

  (void)in_sizes; (void)n_in; (void)out_size; (void)ws_size;

  moe_zero<<<(N_TOKENS * D_MODEL) / 256, 256, 0, stream>>>(out, cnt);
  moe_gate_dispatch<<<N_TOKENS / 4, 128, 0, stream>>>(x, gw, buf, cnt, rtok, rw);
  moe_expert_gemm<<<NUM_EXPERTS * 8, 256, 0, stream>>>(buf, ew, cnt, rtok, rw, out);
}
